// FastAttention_29953101923157
// MI455X (gfx1250) — matrix operand order repaired and checked by reading, not compiled, not run on hardware
//
#include <hip/hip_runtime.h>
#include <hip/hip_bf16.h>

typedef __attribute__((ext_vector_type(16))) __bf16 v16bf;
typedef __attribute__((ext_vector_type(8)))  __bf16 v8bf;
typedef __attribute__((ext_vector_type(4)))  __bf16 v4bf;
typedef __attribute__((ext_vector_type(8)))  float  v8f;
typedef __attribute__((ext_vector_type(4)))  unsigned int v4u;
typedef __attribute__((ext_vector_type(8)))  int    v8i;
typedef __attribute__((ext_vector_type(4)))  int    v4i;

constexpr int B = 2, H = 16, N = 2048, DHEAD = 64;
constexpr int BM = 64;     // query rows per workgroup (16 per wave)
constexpr int BN = 64;     // key cols per iteration
constexpr int VST = 72;    // padded row stride (bf16 elems) of transposed V tile
constexpr float NEG_INF = -3.402823466e38f;   // jnp.finfo(f32).min

struct bf16x16 { v8bf lo, hi; };

// 16 contiguous bf16 (two ds_load_b128)
static __device__ inline v16bf ld16(const __bf16* p) {
    bf16x16 t{ *(const v8bf*)p, *(const v8bf*)(p + 8) };
    return __builtin_bit_cast(v16bf, t);
}
// A-layout run pair: 8 bf16 at p, 8 bf16 at p+16 (two ds_load_b128)
static __device__ inline v16bf ld8_8(const __bf16* p) {
    bf16x16 t{ *(const v8bf*)p, *(const v8bf*)(p + 16) };
    return __builtin_bit_cast(v16bf, t);
}

// ---- TDM: DMA one 64x64 fp32 tile (row-major, stride 64) global -> LDS.
// D# per CDNA5 ISA; tracked by TENSORcnt.
static __device__ inline void tdm_load_tile(const float* gsrc, float* ldst) {
    const unsigned lds = (unsigned)(uintptr_t)ldst;               // LDS byte offset
    const unsigned long long ga = (unsigned long long)(uintptr_t)gsrc;
    v4u g0 = { 1u,                                   // count=1, user descriptor
               lds,                                  // bits[63:32] lds_addr
               (unsigned)ga,                         // global_addr[31:0]
               (unsigned)((ga >> 32) & 0x01FFFFFFu) | 0x80000000u }; // [56:32] | type=2
    v8i g1 = { 0x00020000,        // data_size=2 (4 bytes); wg_mask=0
               64 << 16,          // tensor_dim0=64  (bits 79:48)
               64 << 16,          // tensor_dim1=64  (bits 111:80)
               64 << 16,          // tile_dim0=64    (bits 127:112)
               64,                // tile_dim1=64    (bits 143:128)
               64,                // tensor_dim0_stride=64 (bits 207:160)
               0, 0 };
    v4i gz4 = { 0, 0, 0, 0 };
    v8i gz8 = { 0, 0, 0, 0, 0, 0, 0, 0 };
    __builtin_amdgcn_tensor_load_to_lds(g0, g1, gz4, gz4, gz8, 0);
}

__global__ __launch_bounds__(128)
void fattn_wmma_bf16(const float* __restrict__ q,
                     const float* __restrict__ k,
                     const float* __restrict__ v,
                     const unsigned char* __restrict__ mask,
                     float* __restrict__ out)
{
    __shared__ float  Kraw[2][BN * DHEAD];   // 2 x 16 KB, TDM-filled fp32
    __shared__ float  Vraw[2][BN * DHEAD];   // 2 x 16 KB, TDM-filled fp32
    __shared__ __bf16 Ks[BN * DHEAD];        // 8 KB, row-major [key][d]
    __shared__ __bf16 Vt[DHEAD * VST];       // 9 KB, transposed [d][key], padded

    const int tid  = threadIdx.x;
    const int wave = tid >> 5;
    const int lane = tid & 31;
    const int hf   = lane >> 4;
    const int l16  = lane & 15;

    const int nqb  = N / BM;
    const int qb   = blockIdx.x % nqb;
    const int bh   = blockIdx.x / nqb;
    const int bidx = bh / H;

    const size_t base = (size_t)bh * N * DHEAD;
    const float* Qp = q + base;
    const float* Kp = k + base;
    const float* Vp = v + base;
    const unsigned char* Mp = mask + (size_t)bidx * N;

    // ---- Q^T fragments (B layout): lane = q-column l16, k = d contiguous.
    v16bf qfrag[2];
    {
        const float* qrow = Qp + (size_t)(qb * BM + wave * 16 + l16) * DHEAD;
        #pragma unroll
        for (int kc = 0; kc < 2; ++kc) {
            const float4* f = (const float4*)(qrow + kc * 32 + hf * 16);
            v16bf a;
            #pragma unroll
            for (int i = 0; i < 4; ++i) {
                float4 x = f[i];
                a[4*i+0] = (__bf16)x.x; a[4*i+1] = (__bf16)x.y;
                a[4*i+2] = (__bf16)x.z; a[4*i+3] = (__bf16)x.w;
            }
            qfrag[kc] = a;
        }
    }

    // ---- prologue: wave 0 launches TDM for the first K/V tiles
    if (wave == 0) {
        tdm_load_tile(Kp, &Kraw[0][0]);
        tdm_load_tile(Vp, &Vraw[0][0]);
    }

    v8f o[4] = {};            // O in C layout: row q = r + 8*hf, col d = dt*16+l16
    float m_run = NEG_INF;    // stats for q-row l16 (partner lane = l16+16)
    float l_run = 0.0f;

    for (int j = 0; j < N; j += BN) {
        const int cur = (j / BN) & 1;

        if (wave == 0) __builtin_amdgcn_s_wait_tensorcnt(0);
        __syncthreads();                       // raw[cur] visible to all waves

        // issue next tile's DMA immediately: overlaps convert + 16 WMMAs
        if (wave == 0 && j + BN < N) {
            tdm_load_tile(Kp + (size_t)(j + BN) * DHEAD, &Kraw[cur ^ 1][0]);
            tdm_load_tile(Vp + (size_t)(j + BN) * DHEAD, &Vraw[cur ^ 1][0]);
        }

        // ---- convert raw fp32 -> bf16 tiles
        // K: 8-float units -> one b128 bf16 store (512 units, 4/thread)
        const float4* K4 = (const float4*)&Kraw[cur][0];
        #pragma unroll
        for (int u = 0; u < 4; ++u) {
            const int i8 = tid + u * 128;            // 0..511
            float4 a = K4[2 * i8], b = K4[2 * i8 + 1];
            v8bf kb = { (__bf16)a.x, (__bf16)a.y, (__bf16)a.z, (__bf16)a.w,
                        (__bf16)b.x, (__bf16)b.y, (__bf16)b.z, (__bf16)b.w };
            *(v8bf*)&Ks[i8 * 8] = kb;
        }
        // V: 4x4-block register transpose (256 blocks, 2/thread):
        // 4x ds_load_b128 rows -> 4x ds_store_b64 contiguous-key columns
        const float4* V4 = (const float4*)&Vraw[cur][0];
        #pragma unroll
        for (int u = 0; u < 2; ++u) {
            const int bb   = tid + u * 128;          // 0..255
            const int key0 = (bb >> 4) << 2;
            const int d0   = (bb & 15) << 2;
            float4 r0 = V4[(key0 + 0) * 16 + (d0 >> 2)];
            float4 r1 = V4[(key0 + 1) * 16 + (d0 >> 2)];
            float4 r2 = V4[(key0 + 2) * 16 + (d0 >> 2)];
            float4 r3 = V4[(key0 + 3) * 16 + (d0 >> 2)];
            v4bf c0 = { (__bf16)r0.x, (__bf16)r1.x, (__bf16)r2.x, (__bf16)r3.x };
            v4bf c1 = { (__bf16)r0.y, (__bf16)r1.y, (__bf16)r2.y, (__bf16)r3.y };
            v4bf c2 = { (__bf16)r0.z, (__bf16)r1.z, (__bf16)r2.z, (__bf16)r3.z };
            v4bf c3 = { (__bf16)r0.w, (__bf16)r1.w, (__bf16)r2.w, (__bf16)r3.w };
            *(v4bf*)&Vt[(d0 + 0) * VST + key0] = c0;
            *(v4bf*)&Vt[(d0 + 1) * VST + key0] = c1;
            *(v4bf*)&Vt[(d0 + 2) * VST + key0] = c2;
            *(v4bf*)&Vt[(d0 + 3) * VST + key0] = c3;
        }
        __syncthreads();

        // ---- S^T = K * Q^T (C-layout result == A-layout operand for P*V)
        v8f st[4];
        #pragma unroll
        for (int nt = 0; nt < 4; ++nt) {
            v8f acc = {};
            #pragma unroll
            for (int kc = 0; kc < 2; ++kc) {
                v16bf ka = ld16(&Ks[(nt * 16 + l16) * DHEAD + kc * 32 + hf * 16]);
                acc = __builtin_amdgcn_wmma_f32_16x16x32_bf16(
                          false, ka, false, qfrag[kc], (short)0, acc, false, false);
            }
            st[nt] = acc;
        }

        // ---- mask bits (key = j + nt*16 + r + 8*hf); applied post-exp
        const unsigned long long blo = __ballot(Mp[j + lane] != 0);
        const unsigned long long bhi = __ballot(Mp[j + 32 + lane] != 0);
        const unsigned long long bm  = (bhi << 32) | blo;
        unsigned mb[4];
        #pragma unroll
        for (int nt = 0; nt < 4; ++nt)
            mb[nt] = (unsigned)(bm >> (nt * 16 + 8 * hf)) & 0xFFu;

        // ---- online softmax (masked raw scores may enter the running max:
        // any upper bound keeps online softmax exact -> no pre-exp select)
        float mx = NEG_INF;
        #pragma unroll
        for (int nt = 0; nt < 4; ++nt)
            #pragma unroll
            for (int r = 0; r < 8; ++r) mx = fmaxf(mx, st[nt][r]);
        mx = fmaxf(mx, __shfl_xor(mx, 16, 32));
        const float m_new = fmaxf(m_run, mx);
        const float scale = __expf(m_run - m_new);
        float rsum = 0.0f;
        #pragma unroll
        for (int nt = 0; nt < 4; ++nt)
            #pragma unroll
            for (int r = 0; r < 8; ++r) {
                float e = __expf(st[nt][r] - m_new);
                const unsigned keep = ((mb[nt] >> r) & 1u) - 1u;  // 0 if masked
                e = __uint_as_float(__float_as_uint(e) & keep);   // branch-free zero
                st[nt][r] = e;
                rsum += e;
            }
        rsum += __shfl_xor(rsum, 16, 32);
        l_run = l_run * scale + rsum;
        m_run = m_new;

        #pragma unroll
        for (int r = 0; r < 8; ++r) {
            const float sc = __shfl(scale, r + 8 * hf, 32);
            #pragma unroll
            for (int dt = 0; dt < 4; ++dt) o[dt][r] *= sc;
        }

        // ---- O += P * V, P assembled in-register from S^T
        #pragma unroll
        for (int kc = 0; kc < 2; ++kc) {
            v16bf pa;
            #pragma unroll
            for (int i = 0; i < 8; ++i) {
                pa[i]     = (__bf16)st[2 * kc][i];
                pa[8 + i] = (__bf16)st[2 * kc + 1][i];
            }
            #pragma unroll
            for (int dt = 0; dt < 4; ++dt) {
                v16bf vb = ld8_8(&Vt[(dt * 16 + l16) * VST + kc * 32 + hf * 8]);
                o[dt] = __builtin_amdgcn_wmma_f32_16x16x32_bf16(
                            false, pa, false, vb, (short)0, o[dt], false, false);
            }
        }
        __syncthreads();   // protect Ks/Vt before next convert pass
    }

    // ---- epilogue: normalize and store
    const float linv = 1.0f / l_run;
    #pragma unroll
    for (int r = 0; r < 8; ++r) {
        const float inv = __shfl(linv, r + 8 * hf, 32);
        const int row = qb * BM + wave * 16 + r + 8 * hf;
        #pragma unroll
        for (int dt = 0; dt < 4; ++dt)
            out[base + (size_t)row * DHEAD + dt * 16 + l16] = o[dt][r] * inv;
    }
}

extern "C" void kernel_launch(void* const* d_in, const int* in_sizes, int n_in,
                              void* d_out, int out_size, void* d_ws, size_t ws_size,
                              hipStream_t stream) {
    const float* q = (const float*)d_in[0];
    const float* k = (const float*)d_in[1];
    const float* v = (const float*)d_in[2];
    const unsigned char* mask = (const unsigned char*)d_in[3];  // bool [B,1,1,N]
    float* out = (float*)d_out;

    dim3 grid(B * H * (N / BM));   // 1024 workgroups
    dim3 block(128);               // 4 wave32 waves
    fattn_wmma_bf16<<<grid, block, 0, stream>>>(q, k, v, mask, out);
}
